// MultiHeadAttention_10350871183576
// MI455X (gfx1250) — matrix operand order repaired and checked by reading, not compiled, not run on hardware
//
#include <hip/hip_runtime.h>
#include <hip/hip_bf16.h>

// ---------------- problem constants ----------------
constexpr int Bn = 2;      // batch
constexpr int Sn = 2048;   // sequence
constexpr int Dn = 768;    // d_model
constexpr int Hn = 12;     // heads
constexpr int HD = 64;     // head dim

// ---------------- WMMA types ----------------
typedef __bf16 v16bf __attribute__((ext_vector_type(16)));
typedef __bf16 bf16x8 __attribute__((ext_vector_type(8)));
typedef float  v8f   __attribute__((ext_vector_type(8)));

__device__ __forceinline__ v8f wmma_bf16(v16bf a, v16bf b, v8f c) {
    // v_wmma_f32_16x16x32_bf16 : D = A(16x32) * B(32x16) + C(16x16 f32)
    return __builtin_amdgcn_wmma_f32_16x16x32_bf16(
        /*neg_a=*/false, a, /*neg_b=*/false, b,
        /*c_mod=*/(short)0, c, /*reuse_a=*/false, /*reuse_b=*/false);
}

__device__ __forceinline__ v16bf ld2x8(const __bf16* p0, const __bf16* p1) {
    bf16x8 lo = *(const bf16x8*)p0;   // 16-byte load (global_load_b128 / ds_load_b128)
    bf16x8 hi = *(const bf16x8*)p1;   // 16-byte load
    v16bf r;
#pragma unroll
    for (int i = 0; i < 8; ++i) { r[i] = lo[i]; r[i + 8] = hi[i]; }
    return r;
}

// A fragment (16x32 bf16), memory row-major [row][k], leading dim ld.
// ISA layout: lane<16 -> row=lane, K={kb..kb+7, kb+16..kb+23}, kb=8*(lane>>4)
__device__ __forceinline__ v16bf load_a_frag(const __bf16* base, int row0, int k0,
                                             int ld, int lane) {
    int row = row0 + (lane & 15);
    int kb  = k0 + ((lane >> 4) << 3);
    const __bf16* p = base + row * ld + kb;
    return ld2x8(p, p + 16);
}

// B fragment (32x16 bf16) from TRANSPOSED memory [n][k], leading dim ld.
// ISA layout: lane -> col = n0+(lane&15); K={kb..kb+7, kb+16..kb+23}, kb=k0+8*(lane>>4)
__device__ __forceinline__ v16bf load_bt_frag(const __bf16* base, int n0, int k0,
                                              int ld, int lane) {
    int col = n0 + (lane & 15);
    int kb  = k0 + ((lane >> 4) << 3);
    const __bf16* p = base + col * ld + kb;
    return ld2x8(p, p + 16);
}

// Async DMA of 16B/lane from global into LDS, tracked with ASYNCcnt.
__device__ __forceinline__ void async_b128(const __bf16* gptr, __bf16* lptr) {
    // LDS byte offset within the workgroup allocation = low 32 bits of the
    // generic address of a __shared__ object.
    unsigned laddr = (unsigned)(unsigned long long)(void*)lptr;
    asm volatile("global_load_async_to_lds_b128 %0, %1, off"
                 :: "v"(laddr), "v"(gptr) : "memory");
}

// ---------------- conversion kernels ----------------
__global__ void cvt_f32_bf16(const float* __restrict__ in, __bf16* __restrict__ out, int n) {
    int i = blockIdx.x * 256 + threadIdx.x;
    if (i < n) out[i] = (__bf16)in[i];
}

// [H, D, 64] -> [H, 64, D] bf16 (so GEMM B-frags read contiguous K=d)
__global__ void cvt_w_qkv(const float* __restrict__ W, __bf16* __restrict__ WT) {
    int i = blockIdx.x * 256 + threadIdx.x;
    if (i >= Hn * Dn * HD) return;
    int h = i / (Dn * HD);
    int r = i % (Dn * HD);
    int d = r / HD;
    int e = r % HD;
    WT[(h * HD + e) * Dn + d] = (__bf16)W[i];
}

// [D, D] -> [D, D] transposed bf16
__global__ void cvt_wo(const float* __restrict__ W, __bf16* __restrict__ WT) {
    int i = blockIdx.x * 256 + threadIdx.x;
    if (i >= Dn * Dn) return;
    int d = i / Dn, e = i % Dn;
    WT[e * Dn + d] = (__bf16)W[i];
}

// ---------------- QKV projection ----------------
// grid: (S/128, B*H), block 256 (8 waves). Wave computes 16(s) x 64(e).
// transposed==0 : out is [B,H,S,64]   (Q, K)
// transposed==1 : out is [B,H,64,S]   (V stored transposed for PV B-frags)
__global__ __launch_bounds__(256) void qkv_proj_kernel(
    const __bf16* __restrict__ xb,   // [B,S,D] bf16
    const __bf16* __restrict__ WT,   // [H,64,D] bf16
    const float*  __restrict__ bias, // [H,64] f32
    __bf16* __restrict__ out, int transposed) {
    const int lane  = threadIdx.x & 31;
    const int wave  = threadIdx.x >> 5;
    const int bh    = blockIdx.y;
    const int h     = bh % Hn;
    const int sbase = blockIdx.x * 128 + wave * 16;

    const __bf16* A  = xb + (bh / Hn) * Sn * Dn;
    const __bf16* BT = WT + h * HD * Dn;

    v8f acc[4];
#pragma unroll
    for (int n = 0; n < 4; ++n)
#pragma unroll
        for (int i = 0; i < 8; ++i) acc[n][i] = 0.f;

    for (int k0 = 0; k0 < Dn; k0 += 32) {
        v16bf a = load_a_frag(A, sbase, k0, Dn, lane);
#pragma unroll
        for (int n = 0; n < 4; ++n) {
            v16bf bf = load_bt_frag(BT, n * 16, k0, Dn, lane);
            acc[n] = wmma_bf16(a, bf, acc[n]);
        }
    }

    const int rowbase = (lane >> 4) << 3;
    const int col     = lane & 15;
#pragma unroll
    for (int n = 0; n < 4; ++n) {
        float bv = bias[h * HD + n * 16 + col];
        if (!transposed) {
            __bf16* o = out + (bh * Sn + sbase + rowbase) * HD + n * 16 + col;
#pragma unroll
            for (int r = 0; r < 8; ++r) o[r * HD] = (__bf16)(acc[n][r] + bv);
        } else {
            __bf16* o = out + (bh * HD + n * 16 + col) * Sn + sbase + rowbase;
#pragma unroll
            for (int r = 0; r < 8; ++r) o[r] = (__bf16)(acc[n][r] + bv);
        }
    }
}

// ---------------- fused flash attention ----------------
// grid: (S/128, B*H), block 256 (8 waves). Wave owns 16 Q rows x full head (64).
// K/V tiles (64 t x 64 e each) are staged into LDS with async DMA, double
// buffered, and shared by all 8 waves (one fetch instead of eight).
__global__ __launch_bounds__(256) void flash_attn_kernel(
    const __bf16* __restrict__ Qb,  // [B,H,S,64]
    const __bf16* __restrict__ Kb,  // [B,H,S,64]
    const __bf16* __restrict__ Vt,  // [B,H,64,S]
    __bf16* __restrict__ attn) {    // [B,S,D] (heads concatenated)
    __shared__ __bf16 kv_lds[2][2][64 * 64];  // [buf][K|V][tile], 32 KB
    __shared__ __bf16 p_lds[8][16 * 64];      // private 2KB P-tile per wave

    const int tid   = threadIdx.x;
    const int lane  = tid & 31;
    const int wave  = tid >> 5;
    const int bh    = blockIdx.y;
    const int b     = bh / Hn;
    const int h     = bh % Hn;
    const int sbase = blockIdx.x * 128 + wave * 16;

    const __bf16* Q = Qb + (bh * Sn + sbase) * HD;
    const __bf16* K = Kb + bh * Sn * HD;
    const __bf16* V = Vt + bh * HD * Sn;

    // cooperative async stage of one 64x64 K tile + one 64x64 V tile
    auto stage_tile = [&](int t0, int buf) {
#pragma unroll
        for (int j = 0; j < 2; ++j) {
            int c   = tid + j * 256;       // 512 16B chunks per tile
            int row = c >> 3;              // 0..63
            int off = (c & 7) << 3;        // bf16 elements 0..56
            async_b128(K + (t0 + row) * HD + off, &kv_lds[buf][0][c * 8]);
            async_b128(V + row * Sn + t0 + off,   &kv_lds[buf][1][c * 8]);
        }
    };

    // Q rows held in registers for the whole pass (2 A-frags covering K=0..63)
    const v16bf aQ0 = load_a_frag(Q, 0, 0, HD, lane);
    const v16bf aQ1 = load_a_frag(Q, 0, 32, HD, lane);

    float m[8], l[8];
    v8f o[4];
#pragma unroll
    for (int r = 0; r < 8; ++r) { m[r] = -1.0e30f; l[r] = 0.f; }
#pragma unroll
    for (int n = 0; n < 4; ++n)
#pragma unroll
        for (int i = 0; i < 8; ++i) o[n][i] = 0.f;

    const float c = 0.18033688011112042f;  // log2(e) / sqrt(64)
    const int rowbase = (lane >> 4) << 3;
    const int colid   = lane & 15;
    __bf16* pl = &p_lds[wave][0];

    constexpr int NT = Sn / 64;  // 32 tiles
    stage_tile(0, 0);            // prologue

    for (int it = 0; it < NT; ++it) {
        asm volatile("s_wait_asynccnt 0x0" ::: "memory");  // tile `it` landed
        __syncthreads();                                   // visible to all waves
        if (it + 1 < NT) stage_tile((it + 1) * 64, (it + 1) & 1);  // overlap DMA

        const __bf16* kb = &kv_lds[it & 1][0][0];
        const __bf16* vb = &kv_lds[it & 1][1][0];

        // ---- scores: S = Q * K^T (4 sub-tiles of 16 cols, K-depth 64) ----
        v8f sc[4];
#pragma unroll
        for (int n = 0; n < 4; ++n) {
            v16bf b0 = load_bt_frag(kb, n * 16, 0, HD, lane);
            v16bf b1 = load_bt_frag(kb, n * 16, 32, HD, lane);
            v8f s;
#pragma unroll
            for (int i = 0; i < 8; ++i) s[i] = 0.f;
            s = wmma_bf16(aQ0, b0, s);
            s = wmma_bf16(aQ1, b1, s);
            sc[n] = s;
        }
        // ---- online softmax (element r <-> row, 16-lane row reductions) ----
#pragma unroll
        for (int r = 0; r < 8; ++r) {
            float tm = fmaxf(fmaxf(sc[0][r], sc[1][r]), fmaxf(sc[2][r], sc[3][r]));
#pragma unroll
            for (int off = 1; off < 16; off <<= 1)
                tm = fmaxf(tm, __shfl_xor(tm, off, 16));
            float nm    = fmaxf(m[r], tm);
            float alpha = exp2f((m[r] - nm) * c);
            m[r]        = nm;
            float sum = 0.f;
#pragma unroll
            for (int n = 0; n < 4; ++n) {
                float p = exp2f((sc[n][r] - nm) * c);
                sc[n][r] = p;
                sum += p;
            }
#pragma unroll
            for (int off = 1; off < 16; off <<= 1)
                sum += __shfl_xor(sum, off, 16);
            l[r] = l[r] * alpha + sum;
#pragma unroll
            for (int n = 0; n < 4; ++n) o[n][r] *= alpha;
        }
        // ---- transpose P through LDS (C-layout -> A-layout) ----
#pragma unroll
        for (int r = 0; r < 8; ++r)
#pragma unroll
            for (int n = 0; n < 4; ++n)
                pl[(rowbase + r) * 64 + n * 16 + colid] = (__bf16)sc[n][r];
        asm volatile("s_wait_dscnt 0" ::: "memory");
        v16bf aP0 = load_a_frag(pl, 0, 0, 64, lane);
        v16bf aP1 = load_a_frag(pl, 0, 32, 64, lane);
        // ---- O += P * V  (B-frags from the LDS-resident transposed V) ----
#pragma unroll
        for (int n = 0; n < 4; ++n) {
            v16bf bv0 = load_bt_frag(vb, n * 16, 0, 64, lane);
            v16bf bv1 = load_bt_frag(vb, n * 16, 32, 64, lane);
            o[n] = wmma_bf16(aP0, bv0, o[n]);
            o[n] = wmma_bf16(aP1, bv1, o[n]);
        }
        asm volatile("s_wait_dscnt 0" ::: "memory");  // WAR fence on LDS P tile
    }
    // ---- normalize and emit concatenated-head layout [B,S,D] ----
#pragma unroll
    for (int r = 0; r < 8; ++r) {
        float inv = 1.f / l[r];
        int row   = sbase + rowbase + r;
#pragma unroll
        for (int n = 0; n < 4; ++n)
            attn[(b * Sn + row) * Dn + h * HD + n * 16 + colid] = (__bf16)(o[n][r] * inv);
    }
}

// ---------------- output projection ----------------
// grid: (B*S/128, D/64), block 256 (8 waves). Wave: 16(m) x 64(n).
__global__ __launch_bounds__(256) void out_proj_kernel(
    const __bf16* __restrict__ attn,  // [B*S, D] bf16
    const __bf16* __restrict__ WoT,   // [D, D] transposed bf16
    const float*  __restrict__ bo,    // [D]
    float* __restrict__ out) {        // [B*S, D] f32
    const int lane  = threadIdx.x & 31;
    const int wave  = threadIdx.x >> 5;
    const int mbase = blockIdx.x * 128 + wave * 16;
    const int nbase = blockIdx.y * 64;

    v8f acc[4];
#pragma unroll
    for (int n = 0; n < 4; ++n)
#pragma unroll
        for (int i = 0; i < 8; ++i) acc[n][i] = 0.f;

    for (int k0 = 0; k0 < Dn; k0 += 32) {
        v16bf a = load_a_frag(attn, mbase, k0, Dn, lane);
#pragma unroll
        for (int n = 0; n < 4; ++n) {
            v16bf bf = load_bt_frag(WoT, nbase + n * 16, k0, Dn, lane);
            acc[n] = wmma_bf16(a, bf, acc[n]);
        }
    }
    const int rowbase = (lane >> 4) << 3;
    const int col     = lane & 15;
#pragma unroll
    for (int n = 0; n < 4; ++n) {
        float bv = bo[nbase + n * 16 + col];
#pragma unroll
        for (int r = 0; r < 8; ++r)
            out[(mbase + rowbase + r) * Dn + nbase + n * 16 + col] = acc[n][r] + bv;
    }
}

// ---------------- host-side orchestration ----------------
extern "C" void kernel_launch(void* const* d_in, const int* in_sizes, int n_in,
                              void* d_out, int out_size, void* d_ws, size_t ws_size,
                              hipStream_t stream) {
    const float* x  = (const float*)d_in[0];
    const float* Wq = (const float*)d_in[1];
    const float* Wk = (const float*)d_in[2];
    const float* Wv = (const float*)d_in[3];
    const float* bq = (const float*)d_in[4];
    const float* bk = (const float*)d_in[5];
    const float* bv = (const float*)d_in[6];
    const float* Wo = (const float*)d_in[7];
    const float* bo = (const float*)d_in[8];
    float* out = (float*)d_out;

    // workspace carve-up (bf16 buffers)
    char* ws = (char*)d_ws;
    size_t off = 0;
    auto carve = [&](size_t elems) {
        __bf16* p = (__bf16*)(ws + off);
        off += ((elems * sizeof(__bf16) + 255) & ~size_t(255));
        return p;
    };
    __bf16* xb   = carve((size_t)Bn * Sn * Dn);   // 6.29 MB
    __bf16* WqT  = carve((size_t)Hn * HD * Dn);   // 1.18 MB
    __bf16* WkT  = carve((size_t)Hn * HD * Dn);
    __bf16* WvT  = carve((size_t)Hn * HD * Dn);
    __bf16* WoT  = carve((size_t)Dn * Dn);
    __bf16* Qb   = carve((size_t)Bn * Hn * Sn * HD);
    __bf16* Kb   = carve((size_t)Bn * Hn * Sn * HD);
    __bf16* Vt   = carve((size_t)Bn * Hn * HD * Sn);
    __bf16* attn = carve((size_t)Bn * Sn * Dn);
    (void)ws_size; (void)in_sizes; (void)n_in; (void)out_size;

    // phase 0: convert / transpose to bf16
    {
        int n = Bn * Sn * Dn;
        cvt_f32_bf16<<<(n + 255) / 256, 256, 0, stream>>>(x, xb, n);
        int nw = Hn * Dn * HD;
        cvt_w_qkv<<<(nw + 255) / 256, 256, 0, stream>>>(Wq, WqT);
        cvt_w_qkv<<<(nw + 255) / 256, 256, 0, stream>>>(Wk, WkT);
        cvt_w_qkv<<<(nw + 255) / 256, 256, 0, stream>>>(Wv, WvT);
        int no = Dn * Dn;
        cvt_wo<<<(no + 255) / 256, 256, 0, stream>>>(Wo, WoT);
    }
    // phase 1: QKV projections (V stored transposed)
    {
        dim3 grid(Sn / 128, Bn * Hn);
        qkv_proj_kernel<<<grid, 256, 0, stream>>>(xb, WqT, bq, Qb, 0);
        qkv_proj_kernel<<<grid, 256, 0, stream>>>(xb, WkT, bk, Kb, 0);
        qkv_proj_kernel<<<grid, 256, 0, stream>>>(xb, WvT, bv, Vt, 1);
    }
    // phase 2: fused flash attention (async-LDS staged K/V)
    {
        dim3 grid(Sn / 128, Bn * Hn);
        flash_attn_kernel<<<grid, 256, 0, stream>>>(Qb, Kb, Vt, attn);
    }
    // phase 3: output projection
    {
        dim3 grid((Bn * Sn) / 128, Dn / 64);
        out_proj_kernel<<<grid, 256, 0, stream>>>(attn, WoT, bo, out);
    }
}
